// BatchReformer_22728966930488
// MI455X (gfx1250) — matrix operand order repaired and checked by reading, not compiled, not run on hardware
//
#include <hip/hip_runtime.h>

// ---------------------------------------------------------------------------
// Reformer LSH attention for MI455X (gfx1250, wave32, WMMA).
// One 256-thread workgroup (8 wave32) per (b*h, chunk) problem: T=1024, D=64,
// 4 hash rounds, 64 sort-chunks of 64 q x 128 kv attention windows.
//
// qk rows are stored in LDS *pre-normalized* (unit L2 norm) with the norms
// kept separately: dots_ij = ||q_i|| * (qn_i . qn_j), so both WMMA operands
// of QK^T are plain contiguous LDS loads and the normalization folds into the
// existing per-element output scale. Bucketing (argmax of qk.rot) is
// invariant to positive row scaling, so it is computed from qn directly.
// K-accumulation uses the chained WMMA form (C in, D out): the chain hazard
// NOPs are cheaper than the register pressure of split accumulators.
// ---------------------------------------------------------------------------

typedef __attribute__((ext_vector_type(16))) _Float16 v16h;
typedef __attribute__((ext_vector_type(8)))  float    v8f;

#define T_LEN      1024
#define DHEAD      64
#define NHASH      4
#define NBUCK      16      // buckets per hash round (total 64)
#define NCH        64      // sort chunks (of 64 rows) per problem
#define MASKV      (-50000.0f)
#define SCALE      0.125f  // D^-0.5

// ---- LDS layout (bytes) ----------------------------------------------------
#define OFF_QN    0                         // f16 [1024][64]  = 131072 (unit rows)
#define OFF_V     131072                    // f16 [1024][64]  = 131072
#define OFF_NRM   262144                    // f32 [1024]      = 4096   (row norms)
#define OFF_STK   266240                    // u16 [4096]      = 8192   (h*1024+t, sorted order)
#define OFF_LSE   274432                    // f32 [4096]      = 16384  (indexed by h*1024+t)
#define OFF_TOT   290816                    // f32 [1024]      = 4096   (logsumexp over hashes)
#define OFF_HIST  294912                    // u32 [64]        = 256
#define OFF_PREF  295168                    // u32 [68]        = 288 (padded)
#define OFF_UNION 295456
//   union A: bucket u8[4096] (=4096) + rot f32[2048] (=8192)   -> 12288
//   union B: pstage f16[8 waves][16*32]                        -> 8192
#define OFF_BUCK  OFF_UNION
#define OFF_ROT   (OFF_UNION + 4096)
#define OFF_PST   OFF_UNION
#define SMEM_BYTES (OFF_UNION + 12288)      // 307744 < 320 KB

// A-fragment K mapping for 16-bit 16x32 A matrix (ISA 7.12.2):
// lanes 0-15 hold K 0..7 (v0..3) and 16..23 (v4..7); lanes 16-31 are +8.
__device__ __forceinline__ int a_kmap(int j, int laneHi) {
    int i  = j >> 1;
    int kb = (i < 4) ? (i << 1) : (16 + ((i - 4) << 1));
    return kb + (j & 1) + (laneHi ? 8 : 0);
}
// B-fragment K mapping for 16-bit 32x16 B matrix: same K order as A
// (lanes 0-15 K 0..7 and 16..23; lanes 16-31 are +8).
__device__ __forceinline__ int b_kmap(int j, int laneHi) {
    return ((j < 8) ? j : (16 + (j - 8))) + (laneHi ? 8 : 0);
}

// ---------------------------------------------------------------------------
// One 16-row (m-tile) slice of one 64x128 attention window.
// PASSB=false: compute dots + row LSE only (stored to s_lse[h*1024+t]).
// PASSB=true : recompute dots, weight p by cross-hash softmax, PV via WMMA,
//              scatter/accumulate into the output slice.
// ---------------------------------------------------------------------------
template<bool PASSB>
__device__ __forceinline__ void attn_tile(
    const _Float16* s_qn, const _Float16* s_v, const float* s_nrm,
    const unsigned short* s_stk, float* s_lse, const float* s_tot,
    _Float16* s_pst_wave, int c, int mtile, int lane,
    bool accumulate, float* out_base)
{
    const int laneHi = lane >> 4;
    const int ln     = lane & 15;
    const int cprev  = (c + NCH - 1) & (NCH - 1);

    // sticker values + query-row scale for the 8 rows this lane half owns
    int sqr[8], tqr[8];
    float nqs[8];
#pragma unroll
    for (int vk = 0; vk < 8; ++vk) {
        int qi  = c * 64 + mtile * 16 + vk + laneHi * 8;
        sqr[vk] = s_stk[qi];
        tqr[vk] = sqr[vk] & (T_LEN - 1);
        nqs[vk] = SCALE * s_nrm[tqr[vk]];   // fold ||q_i|| * D^-0.5 into output
    }

    // A fragments: normalized rows, row m = ln, K = d (two 32-chunks)
    int tqa = s_stk[c * 64 + mtile * 16 + ln] & (T_LEN - 1);
    const _Float16* qa = s_qn + tqa * DHEAD;
    v16h a0, a1;
#pragma unroll
    for (int j = 0; j < 16; ++j) {
        int K = a_kmap(j, laneHi);
        a0[j] = qa[K];
        a1[j] = qa[32 + K];
    }

    v8f acc[8];
#pragma unroll
    for (int nt = 0; nt < 8; ++nt) {
        // B column n = ln -> kv index; keys are the same normalized rows
        int kvi  = nt * 16 + ln;
        int srcc = (kvi < 64) ? c : cprev;
        int srci = (kvi < 64) ? kvi : (kvi - 64);
        int tk   = s_stk[srcc * 64 + srci] & (T_LEN - 1);
        const _Float16* kb = s_qn + tk * DHEAD;
        v16h b0, b1;
#pragma unroll
        for (int j = 0; j < 16; ++j) {
            int K = b_kmap(j, laneHi);
            b0[j] = kb[K];
            b1[j] = kb[32 + K];
        }
        v8f z = {0.f, 0.f, 0.f, 0.f, 0.f, 0.f, 0.f, 0.f};
        v8f t0 = __builtin_amdgcn_wmma_f32_16x16x32_f16(false, a0, false, b0,
                                                        (short)0, z,  false, false);
        acc[nt] = __builtin_amdgcn_wmma_f32_16x16x32_f16(false, a1, false, b1,
                                                         (short)0, t0, false, false);
        // scale by ||q_i||*D^-0.5 + self-mask (t_q == t_k)
#pragma unroll
        for (int vk = 0; vk < 8; ++vk) {
            float d = acc[nt][vk] * nqs[vk];
            acc[nt][vk] = (tqr[vk] == tk) ? MASKV : d;
        }
    }

    // row softmax over 128 keys: reduce over 8 n-tiles then 16 lanes of half
    float rowm[8], rows[8];
#pragma unroll
    for (int vk = 0; vk < 8; ++vk) {
        float m = -3.0e38f;
#pragma unroll
        for (int nt = 0; nt < 8; ++nt) m = fmaxf(m, acc[nt][vk]);
#pragma unroll
        for (int off = 1; off < 16; off <<= 1) m = fmaxf(m, __shfl_xor(m, off, 32));
        float s = 0.f;
#pragma unroll
        for (int nt = 0; nt < 8; ++nt) {
            float e = __expf(acc[nt][vk] - m);
            acc[nt][vk] = e;
            s += e;
        }
#pragma unroll
        for (int off = 1; off < 16; off <<= 1) s += __shfl_xor(s, off, 32);
        rowm[vk] = m;
        rows[vk] = s;
    }

    if (!PASSB) {
        if (ln == 0) {
#pragma unroll
            for (int vk = 0; vk < 8; ++vk)
                s_lse[sqr[vk]] = rowm[vk] + __logf(rows[vk]);
        }
        return;
    }

    // ---- PASS B ----
    // p' = exp(d - lse) * exp(lse - total) = acc * f, f = exp(lse-total)/sum
    float f[8];
#pragma unroll
    for (int vk = 0; vk < 8; ++vk) {
        float lse = rowm[vk] + __logf(rows[vk]);
        float w   = __expf(lse - s_tot[tqr[vk]]);
        f[vk]     = w / rows[vk];
    }

    v8f acc4[4];
#pragma unroll
    for (int dt = 0; dt < 4; ++dt)
        acc4[dt] = (v8f){0.f, 0.f, 0.f, 0.f, 0.f, 0.f, 0.f, 0.f};

    for (int kt = 0; kt < 4; ++kt) {
        // stage this 32-key window of p' as f16 [16][32] (C-layout -> LDS)
#pragma unroll
        for (int half2 = 0; half2 < 2; ++half2) {
            int nt = kt * 2 + half2;
#pragma unroll
            for (int vk = 0; vk < 8; ++vk) {
                int mrow = vk + laneHi * 8;
                int col  = half2 * 16 + ln;
                s_pst_wave[mrow * 32 + col] = (_Float16)(acc[nt][vk] * f[vk]);
            }
        }
        __syncthreads();   // uniform: all waves run identical trip counts

        // A fragment of p' (row = ln)
        v16h ap;
#pragma unroll
        for (int j = 0; j < 16; ++j) {
            int K = a_kmap(j, laneHi);
            ap[j] = s_pst_wave[ln * 32 + K];
        }
        // key t's for this lane's K list in this window
        int tkw[16];
#pragma unroll
        for (int j = 0; j < 16; ++j) {
            int K    = b_kmap(j, laneHi);
            int kvi  = kt * 32 + K;
            int srcc = (kvi < 64) ? c : cprev;
            int srci = (kvi < 64) ? kvi : (kvi - 64);
            tkw[j]   = s_stk[srcc * 64 + srci] & (T_LEN - 1);
        }
        // B fragments of bv (col n = ln -> d); 4 independent dt chains
        // interleave to cover the accumulator-chain hazard window.
#pragma unroll
        for (int dt = 0; dt < 4; ++dt) {
            int d = dt * 16 + ln;
            v16h bv;
#pragma unroll
            for (int j = 0; j < 16; ++j)
                bv[j] = s_v[tkw[j] * DHEAD + d];
            acc4[dt] = __builtin_amdgcn_wmma_f32_16x16x32_f16(false, ap, false, bv,
                                                              (short)0, acc4[dt],
                                                              false, false);
        }
        __syncthreads();   // WAR on the staging buffer
    }

    // scatter weighted output rows (unsort): row m -> original t
#pragma unroll
    for (int dt = 0; dt < 4; ++dt) {
        int d = dt * 16 + ln;
#pragma unroll
        for (int vk = 0; vk < 8; ++vk) {
            float vval = acc4[dt][vk];
            float* p = out_base + (size_t)tqr[vk] * DHEAD + d;
            if (accumulate) *p = *p + vval;
            else            *p = vval;
        }
    }
}

// ---------------------------------------------------------------------------
__launch_bounds__(256, 1)
__global__ void lsh_attn_kernel(const float* __restrict__ query,
                                const float* __restrict__ value,
                                const float* __restrict__ rotations,
                                float* __restrict__ out)
{
    extern __shared__ char smem[];
    _Float16*       s_qn  = (_Float16*)      (smem + OFF_QN);
    _Float16*       s_v   = (_Float16*)      (smem + OFF_V);
    float*          s_nrm = (float*)         (smem + OFF_NRM);
    unsigned short* s_stk = (unsigned short*)(smem + OFF_STK);
    float*          s_lse = (float*)         (smem + OFF_LSE);
    float*          s_tot = (float*)         (smem + OFF_TOT);
    unsigned*       s_hist= (unsigned*)      (smem + OFF_HIST);
    unsigned*       s_pref= (unsigned*)      (smem + OFF_PREF);
    unsigned char*  s_buck= (unsigned char*) (smem + OFF_BUCK);
    float*          s_rot = (float*)         (smem + OFF_ROT);
    _Float16*       s_pst = (_Float16*)      (smem + OFF_PST);

    const int tid  = threadIdx.x;
    const int lane = tid & 31;
    const int wv   = tid >> 5;

    const int prob  = blockIdx.x;        // 0..255
    const int chunk = prob & 3;
    const int bh    = prob >> 2;         // 0..63
    const int b     = bh >> 4;
    const int head  = bh & 15;

    const float* qbase = query +
        ((size_t)(b * 4096 + chunk * T_LEN)) * 1024 + (size_t)head * DHEAD;
    const float* vbase = value +
        ((size_t)(b * T_LEN)) * 1024 + (size_t)head * DHEAD;
    const float* rbase = rotations + (size_t)chunk * (DHEAD * NHASH * 8);
    float* obase = out +
        ((size_t)(b * 16 + head) * 4096 + (size_t)chunk * T_LEN) * DHEAD;

    // ---- Phase 0: stage normalized qk + norms, v (f16); load rotations ----
    if (tid < 64) s_hist[tid] = 0u;
    for (int t = wv; t < T_LEN; t += 8) {
        const float* qr = qbase + (size_t)t * 1024;
        const float* vr = vbase + (size_t)t * 1024;
        int d0 = lane * 2;
        float x0 = qr[d0], x1 = qr[d0 + 1];
        float y0 = vr[d0], y1 = vr[d0 + 1];
        float ss = x0 * x0 + x1 * x1;
#pragma unroll
        for (int off = 16; off > 0; off >>= 1) ss += __shfl_xor(ss, off, 32);
        float nrm = fmaxf(sqrtf(ss), 1e-12f);   // all lanes hold full reduction
        float inv = 1.0f / nrm;
        s_qn[t * DHEAD + d0]     = (_Float16)(x0 * inv);
        s_qn[t * DHEAD + d0 + 1] = (_Float16)(x1 * inv);
        s_v [t * DHEAD + d0]     = (_Float16)y0;
        s_v [t * DHEAD + d0 + 1] = (_Float16)y1;
        if (lane == 0) s_nrm[t] = nrm;
    }
    for (int i = tid; i < DHEAD * NHASH * 8; i += 256) s_rot[i] = rbase[i];
    __syncthreads();

    // ---- Phase 1: LSH bucketing (argmax over [r, -r]) ----------------------
    // argmax(qk.rot) == argmax(qn.rot): invariant to positive row scaling.
    for (int idx = tid; idx < NHASH * T_LEN; idx += 256) {
        int h = idx >> 10, t = idx & (T_LEN - 1);
        float r[8];
#pragma unroll
        for (int j = 0; j < 8; ++j) r[j] = 0.f;
        for (int d = 0; d < DHEAD; ++d) {
            float q = (float)s_qn[t * DHEAD + d];
            const float* rp = s_rot + d * (NHASH * 8) + h * 8;
#pragma unroll
            for (int j = 0; j < 8; ++j) r[j] += q * rp[j];
        }
        int best = 0; float bv = r[0];
#pragma unroll
        for (int j = 1; j < 8; ++j)  if (r[j]  > bv) { bv = r[j];  best = j; }
#pragma unroll
        for (int j = 0; j < 8; ++j)  if (-r[j] > bv) { bv = -r[j]; best = 8 + j; }
        int bucket = h * NBUCK + best;
        s_buck[idx] = (unsigned char)bucket;
        atomicAdd(&s_hist[bucket], 1u);
    }
    __syncthreads();

    // ---- Phase 2: exclusive prefix over 64 buckets -------------------------
    if (tid == 0) {
        unsigned run = 0;
        for (int i = 0; i < 64; ++i) { s_pref[i] = run; run += s_hist[i]; }
        s_pref[64] = run;
    }
    __syncthreads();
    if (tid < 64) s_hist[tid] = 0u;   // reuse as running counters
    __syncthreads();

    // ---- Phase 3: stable counting-sort scatter (1 wave per hash round) ----
    if (wv < NHASH) {
        const int h = wv;
        for (int g = 0; g < T_LEN / 32; ++g) {   // t ascending => stable
            int t = g * 32 + lane;
            int bucket = s_buck[h * T_LEN + t];
            int bl = bucket & (NBUCK - 1);
            unsigned m = 0u;
#pragma unroll
            for (int bb = 0; bb < NBUCK; ++bb) {
                unsigned mm = (unsigned)__ballot(bl == bb);
                if (bl == bb) m = mm;
            }
            int rank   = __popc(m & ((1u << lane) - 1u));
            int leader = __builtin_ctz(m);
            unsigned base = 0u;
            if (lane == leader) base = atomicAdd(&s_hist[bucket], (unsigned)__popc(m));
            base = __shfl(base, leader, 32);
            unsigned pos = s_pref[bucket] + base + (unsigned)rank;
            s_stk[pos] = (unsigned short)(h * T_LEN + t);
        }
    }
    __syncthreads();

    // ---- Pass A: dots + per-row LSE for all 64 windows ---------------------
    for (int cc = 0; cc < 8; ++cc) {
        int c = cc * 8 + wv;
        for (int mtile = 0; mtile < 4; ++mtile)
            attn_tile<false>(s_qn, s_v, s_nrm, s_stk, s_lse, s_tot,
                             s_pst + wv * 512, c, mtile, lane, false, obase);
    }
    __syncthreads();

    // ---- cross-hash logsumexp per t ---------------------------------------
    for (int t = tid; t < T_LEN; t += 256) {
        float l0 = s_lse[t], l1 = s_lse[T_LEN + t];
        float l2 = s_lse[2 * T_LEN + t], l3 = s_lse[3 * T_LEN + t];
        float m = fmaxf(fmaxf(l0, l1), fmaxf(l2, l3));
        float s = __expf(l0 - m) + __expf(l1 - m) + __expf(l2 - m) + __expf(l3 - m);
        s_tot[t] = m + __logf(s);
    }
    __syncthreads();

    // ---- Pass B: weighted PV, accumulate per hash round (no races) --------
    for (int h = 0; h < NHASH; ++h) {
        for (int sub = 0; sub < 2; ++sub) {
            int c = h * 16 + sub * 8 + wv;
            for (int mtile = 0; mtile < 4; ++mtile)
                attn_tile<true>(s_qn, s_v, s_nrm, s_stk, s_lse, s_tot,
                                s_pst + wv * 512, c, mtile, lane,
                                /*accumulate=*/(h > 0), obase);
        }
        __threadfence();
        __syncthreads();
    }
}

// ---------------------------------------------------------------------------
extern "C" void kernel_launch(void* const* d_in, const int* in_sizes, int n_in,
                              void* d_out, int out_size, void* d_ws, size_t ws_size,
                              hipStream_t stream)
{
    (void)in_sizes; (void)n_in; (void)out_size; (void)d_ws; (void)ws_size;
    const float* query     = (const float*)d_in[0];
    // d_in[1] ("key") is unused by the reference (shared-QK attention)
    const float* value     = (const float*)d_in[2];
    const float* rotations = (const float*)d_in[3];
    float* out = (float*)d_out;

    static_assert(SMEM_BYTES <= 320 * 1024, "LDS budget exceeded");
    hipFuncSetAttribute((const void*)lsh_attn_kernel,
                        hipFuncAttributeMaxDynamicSharedMemorySize, SMEM_BYTES);

    dim3 grid(256);   // (B*H) * N_CHUNKS problems
    dim3 block(256);  // 8 wave32
    lsh_attn_kernel<<<grid, block, SMEM_BYTES, stream>>>(query, value, rotations, out);
}
